// GCN_5222680232345
// MI455X (gfx1250) — matrix operand order repaired and checked by reading, not compiled, not run on hardware
//
#include <hip/hip_runtime.h>
#include <hip/hip_bf16.h>
#include <math.h>

typedef __attribute__((ext_vector_type(16))) __bf16 v16bf;
typedef __attribute__((ext_vector_type(8)))  float  v8f;

#define F_IN  128
#define F_HID 64
#define F_OUT 40

// -------------------------------------------------------------------------
// GEMM1: XW = X[n,128] @ W1[128,64]   (bf16 WMMA, f32 accumulate)
// block = 128 threads = 4 wave32; block covers 64 rows; wave covers 16 rows.
// Tiles are staged into LDS pre-packed in WMMA fragment layout so each
// fragment is ONE 32-byte LDS load (2x ds_load_b128) per lane.
// -------------------------------------------------------------------------
__global__ void __launch_bounds__(128)
gcn_gemm1(const float* __restrict__ X, const float* __restrict__ W1,
          float* __restrict__ XW, int n)
{
    __shared__ v16bf Ap[4][4][32];   // [wave][ktile][lane] : 16 KB
    __shared__ v16bf Bp[4][4][32];   // [ktile][ntile][lane] : 16 KB
    const int rowBase = blockIdx.x * 64;
    const int tid = threadIdx.x;

    // stage X tile (f32 -> bf16) directly into A-fragment layout.
    // 16-bit A 16x32: lane-half h=(kk>>3)&1, elem pos=(kk&7)+8*(kk>>4)
    for (int idx = tid; idx < 64 * F_IN; idx += 128) {
        int r = idx >> 7, k = idx & 127;
        int gr = rowBase + r;
        float v = (gr < n) ? X[(size_t)gr * F_IN + k] : 0.0f;
        int w = r >> 4, l15 = r & 15;
        int kt = k >> 5, kk = k & 31;
        int h   = (kk >> 3) & 1;
        int pos = (kk & 7) + ((kk >> 4) << 3);
        ((__bf16*)&Ap[w][kt][h * 16 + l15])[pos] = (__bf16)v;
    }
    // stage W1 into B-fragment layout.
    // 16-bit B 32x16: lane-half hb=(kk>>3)&1, elem pos=(kk&7)+8*(kk>>4)
    for (int idx = tid; idx < F_IN * F_HID; idx += 128) {
        int k = idx >> 6, c = idx & 63;
        float v = W1[idx];
        int kt = k >> 5, kk = k & 31;
        int nt = c >> 4, l15 = c & 15;
        ((__bf16*)&Bp[kt][nt][((kk >> 3) & 1) * 16 + l15])[(kk & 7) + ((kk >> 4) << 3)] = (__bf16)v;
    }
    __syncthreads();

    const int lane = tid & 31;
    const int wave = tid >> 5;
    const int half = lane >> 4;
    const int l15  = lane & 15;

    v8f acc[4];
    #pragma unroll
    for (int t = 0; t < 4; ++t)
        #pragma unroll
        for (int i = 0; i < 8; ++i) acc[t][i] = 0.0f;

    #pragma unroll
    for (int kt = 0; kt < 4; ++kt) {
        v16bf a = Ap[wave][kt][lane];
        #pragma unroll
        for (int t = 0; t < 4; ++t) {
            v16bf b = Bp[kt][t][lane];
            acc[t] = __builtin_amdgcn_wmma_f32_16x16x32_bf16(
                false, a, false, b, (short)0, acc[t], false, false);
        }
    }
    // C/D layout: VGPR i -> row i (+8 for upper lane half), col = lane&15
    #pragma unroll
    for (int t = 0; t < 4; ++t) {
        const int n0 = t * 16;
        #pragma unroll
        for (int i = 0; i < 8; ++i) {
            int r = rowBase + wave * 16 + half * 8 + i;
            if (r < n) XW[(size_t)r * F_HID + n0 + l15] = acc[t][i];
        }
    }
}

// -------------------------------------------------------------------------
// GEMM2: HW = H[n,64] @ W2[64,40]  (N padded to 48 = 3 WMMA tiles)
// -------------------------------------------------------------------------
__global__ void __launch_bounds__(128)
gcn_gemm2(const float* __restrict__ H, const float* __restrict__ W2,
          float* __restrict__ HW, int n)
{
    __shared__ v16bf Ap[4][2][32];   // [wave][ktile][lane] : 8 KB
    __shared__ v16bf Bp[2][3][32];   // [ktile][ntile][lane] : 6 KB
    const int rowBase = blockIdx.x * 64;
    const int tid = threadIdx.x;

    for (int idx = tid; idx < 64 * F_HID; idx += 128) {
        int r = idx >> 6, k = idx & 63;
        int gr = rowBase + r;
        float v = (gr < n) ? H[(size_t)gr * F_HID + k] : 0.0f;
        int w = r >> 4, l15 = r & 15;
        int kt = k >> 5, kk = k & 31;
        int h   = (kk >> 3) & 1;
        int pos = (kk & 7) + ((kk >> 4) << 3);
        ((__bf16*)&Ap[w][kt][h * 16 + l15])[pos] = (__bf16)v;
    }
    for (int idx = tid; idx < F_HID * 48; idx += 128) {
        int k = idx / 48, c = idx % 48;
        float v = (c < F_OUT) ? W2[k * F_OUT + c] : 0.0f;
        int kt = k >> 5, kk = k & 31;
        int nt = c >> 4, l15 = c & 15;
        ((__bf16*)&Bp[kt][nt][((kk >> 3) & 1) * 16 + l15])[(kk & 7) + ((kk >> 4) << 3)] = (__bf16)v;
    }
    __syncthreads();

    const int lane = tid & 31;
    const int wave = tid >> 5;
    const int half = lane >> 4;
    const int l15  = lane & 15;

    v8f acc[3];
    #pragma unroll
    for (int t = 0; t < 3; ++t)
        #pragma unroll
        for (int i = 0; i < 8; ++i) acc[t][i] = 0.0f;

    #pragma unroll
    for (int kt = 0; kt < 2; ++kt) {
        v16bf a = Ap[wave][kt][lane];
        #pragma unroll
        for (int t = 0; t < 3; ++t) {
            v16bf b = Bp[kt][t][lane];
            acc[t] = __builtin_amdgcn_wmma_f32_16x16x32_bf16(
                false, a, false, b, (short)0, acc[t], false, false);
        }
    }
    #pragma unroll
    for (int t = 0; t < 3; ++t) {
        const int n0 = t * 16;
        #pragma unroll
        for (int i = 0; i < 8; ++i) {
            int r = rowBase + wave * 16 + half * 8 + i;
            int c = n0 + l15;
            if (r < n && c < F_OUT) HW[(size_t)r * F_OUT + c] = acc[t][i];
        }
    }
}

// -------------------------------------------------------------------------
// Elementwise / graph kernels
// -------------------------------------------------------------------------
__global__ void set_ones(float* __restrict__ p, int n)
{
    int i = blockIdx.x * blockDim.x + threadIdx.x;
    int st = gridDim.x * blockDim.x;
    for (; i < n; i += st) p[i] = 1.0f;   // self-loop contributes 1 to degree
}

__global__ void deg_accum(const int* __restrict__ dst, float* __restrict__ deg,
                          long long e)
{
    long long i  = (long long)blockIdx.x * blockDim.x + threadIdx.x;
    long long st = (long long)gridDim.x * blockDim.x;
    for (; i < e; i += st) unsafeAtomicAdd(&deg[dst[i]], 1.0f);
}

__global__ void deg_to_dinv(float* __restrict__ deg, int n)
{
    int i = blockIdx.x * blockDim.x + threadIdx.x;
    int st = gridDim.x * blockDim.x;
    for (; i < n; i += st) {
        float d = deg[i];
        deg[i] = (d > 0.0f) ? rsqrtf(d) : 0.0f;
    }
}

template <int F>
__global__ void init_bias(float* __restrict__ out, const float* __restrict__ b, int n)
{
    long long i  = (long long)blockIdx.x * blockDim.x + threadIdx.x;
    long long st = (long long)gridDim.x * blockDim.x;
    long long total = (long long)n * F;
    for (; i < total; i += st) out[i] = b[(int)(i % F)];
}

// one wave per edge: lanes stride the F features (coalesced gather + fp32 atomic scatter)
template <int F>
__global__ void edge_aggregate(const float* __restrict__ in, const float* __restrict__ dinv,
                               const int* __restrict__ src, const int* __restrict__ dst,
                               float* __restrict__ out, long long eEdges, long long eTot)
{
    long long wid = ((long long)blockIdx.x * blockDim.x + threadIdx.x) >> 5;
    long long nw  = ((long long)gridDim.x * blockDim.x) >> 5;
    const int lane = threadIdx.x & 31;
    for (long long e = wid; e < eTot; e += nw) {
        int s, d;
        if (e < eEdges) { s = src[e]; d = dst[e]; }
        else            { s = d = (int)(e - eEdges); }   // self-loop
        float w = dinv[s] * dinv[d];
        #pragma unroll
        for (int f = lane; f < F; f += 32)
            unsafeAtomicAdd(&out[(size_t)d * F + f], w * in[(size_t)s * F + f]);
    }
}

__global__ void relu_kernel(float* __restrict__ p, long long total)
{
    long long i  = (long long)blockIdx.x * blockDim.x + threadIdx.x;
    long long st = (long long)gridDim.x * blockDim.x;
    for (; i < total; i += st) p[i] = fmaxf(p[i], 0.0f);
}

// in-place log_softmax over rows of 40; one wave per row, shuffle reductions
__global__ void log_softmax40(float* __restrict__ z, int n)
{
    long long wid = ((long long)blockIdx.x * blockDim.x + threadIdx.x) >> 5;
    long long nw  = ((long long)gridDim.x * blockDim.x) >> 5;
    const int lane = threadIdx.x & 31;
    for (long long row = wid; row < n; row += nw) {
        float* r = z + row * F_OUT;
        float a = r[lane];                                   // 40 > 32: all lanes valid
        float b = (lane < F_OUT - 32) ? r[32 + lane] : -INFINITY;
        float m = fmaxf(a, b);
        #pragma unroll
        for (int off = 16; off; off >>= 1) m = fmaxf(m, __shfl_xor(m, off, 32));
        float s = __expf(a - m) + ((lane < F_OUT - 32) ? __expf(b - m) : 0.0f);
        #pragma unroll
        for (int off = 16; off; off >>= 1) s += __shfl_xor(s, off, 32);
        float ls = m + __logf(s);
        r[lane] = a - ls;
        if (lane < F_OUT - 32) r[32 + lane] = b - ls;
    }
}

// -------------------------------------------------------------------------
extern "C" void kernel_launch(void* const* d_in, const int* in_sizes, int n_in,
                              void* d_out, int out_size, void* d_ws, size_t ws_size,
                              hipStream_t stream)
{
    const float* X  = (const float*)d_in[0];
    const int*   EI = (const int*)d_in[1];   // [2, E] (src row, dst row)
    const float* W1 = (const float*)d_in[2];
    const float* b1 = (const float*)d_in[3];
    const float* W2 = (const float*)d_in[4];
    const float* b2 = (const float*)d_in[5];

    const int       n = in_sizes[0] / F_IN;
    const long long e = (long long)in_sizes[1] / 2;
    const int* src = EI;
    const int* dst = EI + e;
    const long long eTot = e + n;

    // workspace carve-up (floats)
    float* wsf = (float*)d_ws;
    size_t off = 0;
    float* dinv = wsf + off; off += ((size_t)n + 255) & ~(size_t)255;
    float* xw1  = wsf + off; off += (size_t)n * F_HID;
    float* h    = wsf + off; off += (size_t)n * F_HID;
    float* hw2  = wsf + off; off += (size_t)n * F_OUT;
    float* z    = (float*)d_out;

    const int TB = 256;
    const int gridElemN  = (n + TB - 1) / TB;
    const int gridEdges  = 2048;
    const int gridAgg    = 4096;              // 32768 wave32 waves, grid-stride
    const int gridGemm   = (n + 63) / 64;
    const int gridBias64 = 4096;

    // degree (with self-loop) -> D^{-1/2}
    set_ones<<<gridElemN, TB, 0, stream>>>(dinv, n);
    deg_accum<<<gridEdges, TB, 0, stream>>>(dst, dinv, e);
    deg_to_dinv<<<gridElemN, TB, 0, stream>>>(dinv, n);

    // layer 1
    gcn_gemm1<<<gridGemm, 128, 0, stream>>>(X, W1, xw1, n);
    init_bias<F_HID><<<gridBias64, TB, 0, stream>>>(h, b1, n);
    edge_aggregate<F_HID><<<gridAgg, TB, 0, stream>>>(xw1, dinv, src, dst, h, e, eTot);
    relu_kernel<<<gridBias64, TB, 0, stream>>>(h, (long long)n * F_HID);

    // layer 2
    gcn_gemm2<<<gridGemm, 128, 0, stream>>>(h, W2, hw2, n);
    init_bias<F_OUT><<<gridBias64, TB, 0, stream>>>(z, b2, n);
    edge_aggregate<F_OUT><<<gridAgg, TB, 0, stream>>>(hw2, dinv, src, dst, z, e, eTot);

    // log_softmax rows of 40, in place on d_out
    log_softmax40<<<gridAgg, TB, 0, stream>>>(z, n);
}
